// VanillaEncoder_37795712205000
// MI455X (gfx1250) — matrix operand order repaired and checked by reading, not compiled, not run on hardware
//
#include <hip/hip_runtime.h>
#include <hip/hip_fp16.h>

typedef _Float16 h8  __attribute__((ext_vector_type(8)));
typedef _Float16 v16h __attribute__((ext_vector_type(16)));
typedef float    v8f  __attribute__((ext_vector_type(8)));

#define DMODEL 768
#define NHEAD  12
#define HDIM   64
#define FFDIM  3072
#define NLAT   32
#define SEQ    197
#define BATCH  64
#define MROWS  (BATCH*SEQ)      /* 12608 */
#define QKVN   2304
#define INV_SQRT_D 0.03608439182435161f

// ---------------- CDNA5 async-to-LDS helpers ----------------
__device__ __forceinline__ void async_b128(unsigned lds_off, const void* gaddr) {
  // GLOBAL_LOAD_ASYNC_TO_LDS_B128: per-lane 16B memory -> LDS, tracked by ASYNCcnt
  asm volatile("global_load_async_to_lds_b128 %0, %1, off" :: "v"(lds_off), "v"(gaddr) : "memory");
}
__device__ __forceinline__ void wait_async0() {
  asm volatile("s_wait_asynccnt 0x0" ::: "memory");
}
__device__ __forceinline__ unsigned lds_off(const void* p) {
  // flat shared pointer: low 32 bits are the LDS byte address
  return (unsigned)(size_t)p;
}

// ---------------- WMMA fragment loaders (LDS, f16) ----------------
// A-matrix 16x32 f16 (ISA 7.12.2): lane<16 -> K {kb..kb+7, 16+kb..16+kb+7}, kb = 0 / 8
__device__ __forceinline__ v16h load_a_frag(const _Float16* rowp, int lane) {
  int kb = (lane < 16) ? 0 : 8;
  h8 x0 = *(const h8*)(rowp + kb);
  h8 x1 = *(const h8*)(rowp + kb + 16);
  v16h a;
  for (int e = 0; e < 8; e++) { a[e] = x0[e]; a[e + 8] = x1[e]; }
  return a;
}
// B-matrix 32x16 f16: lane = column n=lane&15, K {kb..kb+7, 16+kb..16+kb+7}, kb = 0 / 8 by lane half.
__device__ __forceinline__ v16h load_b_frag(const _Float16* colp, int lane) {
  int kb = (lane < 16) ? 0 : 8;
  h8 x0 = *(const h8*)(colp + kb);
  h8 x1 = *(const h8*)(colp + kb + 16);
  v16h b;
  for (int e = 0; e < 8; e++) { b[e] = x0[e]; b[e + 8] = x1[e]; }
  return b;
}

// ---------------- weight convert + transpose: w[K][N] f32 -> wt[N][K] f16 ----------------
__global__ __launch_bounds__(256) void k_wt(const float* __restrict__ w,
                                            _Float16* __restrict__ wt,
                                            int K, int N) {
  __shared__ alignas(16) _Float16 tile[64 * 72];
  int t = threadIdx.x;
  int n0 = blockIdx.x * 64, k0 = blockIdx.y * 64;
  {
    int r = t >> 2, c4 = (t & 3) * 16;
    const float4* p = (const float4*)(w + (size_t)(k0 + r) * N + n0 + c4);
    float4 f0 = p[0], f1 = p[1], f2 = p[2], f3 = p[3];
    float fv[16] = {f0.x, f0.y, f0.z, f0.w, f1.x, f1.y, f1.z, f1.w,
                    f2.x, f2.y, f2.z, f2.w, f3.x, f3.y, f3.z, f3.w};
    _Float16* d = tile + r * 72 + c4;
    for (int i = 0; i < 16; i++) d[i] = (_Float16)fv[i];
  }
  __syncthreads();
  {
    int n = t >> 2, k4 = (t & 3) * 16;
    h8 v0, v1;
    for (int i = 0; i < 8; i++) {
      v0[i] = tile[(k4 + i) * 72 + n];
      v1[i] = tile[(k4 + 8 + i) * 72 + n];
    }
    h8* d = (h8*)(wt + (size_t)(n0 + n) * K + k0 + k4);
    d[0] = v0; d[1] = v1;
  }
}

// ---------------- Fusion stage: fused = attn(latents, [x;y]) ----------------
__global__ __launch_bounds__(256) void k_fused(const float* __restrict__ x,
                                               const float* __restrict__ y,
                                               const float* __restrict__ lat,
                                               float* __restrict__ fused) {
  __shared__ float s[2 * SEQ + 2];
  __shared__ float red[256];
  int t = threadIdx.x;
  int b = blockIdx.x >> 5;
  int l = blockIdx.x & 31;
  const float* lr = lat + (size_t)l * DMODEL;
  for (int j = t; j < 2 * SEQ; j += 256) {
    const float* kr = (j < SEQ) ? x + ((size_t)b * SEQ + j) * DMODEL
                                : y + ((size_t)b * SEQ + (j - SEQ)) * DMODEL;
    float acc = 0.f;
    for (int d = 0; d < DMODEL; d++) acc += lr[d] * kr[d];
    s[j] = acc * INV_SQRT_D;
  }
  __syncthreads();
  float lm = -1e30f;
  for (int j = t; j < 2 * SEQ; j += 256) lm = fmaxf(lm, s[j]);
  red[t] = lm; __syncthreads();
  for (int s2 = 128; s2 > 0; s2 >>= 1) { if (t < s2) red[t] = fmaxf(red[t], red[t + s2]); __syncthreads(); }
  float mx = red[0]; __syncthreads();
  float lsum = 0.f;
  for (int j = t; j < 2 * SEQ; j += 256) { float e = __expf(s[j] - mx); s[j] = e; lsum += e; }
  red[t] = lsum; __syncthreads();
  for (int s2 = 128; s2 > 0; s2 >>= 1) { if (t < s2) red[t] += red[t + s2]; __syncthreads(); }
  float inv = 1.f / red[0];
  __syncthreads();
  float* fo = fused + ((size_t)b * NLAT + l) * DMODEL;
  for (int d = t; d < DMODEL; d += 256) {
    float acc = 0.f;
    const float* xp = x + (size_t)b * SEQ * DMODEL + d;
    for (int j = 0; j < SEQ; j++) acc += s[j] * xp[(size_t)j * DMODEL];
    const float* yp = y + (size_t)b * SEQ * DMODEL + d;
    for (int j = 0; j < SEQ; j++) acc += s[SEQ + j] * yp[(size_t)j * DMODEL];
    fo[d] = acc * inv;
  }
}

// ---------------- out = in + scale * attn(in, fused, fused) ----------------
__global__ __launch_bounds__(256) void k_cross(const float* __restrict__ inp,
                                               const float* __restrict__ fused,
                                               const float* __restrict__ scale_p,
                                               float* __restrict__ outp) {
  __shared__ float part[256];
  __shared__ float ps[32];
  __shared__ float inv_s;
  int t = threadIdx.x;
  size_t row = blockIdx.x;            // b*197+n
  int b = (int)(row / SEQ);
  const float* xr = inp + row * DMODEL;
  const float* fb = fused + (size_t)b * NLAT * DMODEL;
  int l = t >> 3, seg = t & 7;
  const float* fr = fb + (size_t)l * DMODEL + seg * 96;
  const float* xs = xr + seg * 96;
  float acc = 0.f;
  for (int d = 0; d < 96; d++) acc += xs[d] * fr[d];
  part[t] = acc;
  __syncthreads();
  if (t < 32) {
    float s = 0.f;
    for (int e = 0; e < 8; e++) s += part[t * 8 + e];
    ps[t] = s * INV_SQRT_D;
  }
  __syncthreads();
  if (t == 0) {
    float mx = -1e30f;
    for (int i = 0; i < 32; i++) mx = fmaxf(mx, ps[i]);
    float sm = 0.f;
    for (int i = 0; i < 32; i++) { float e = __expf(ps[i] - mx); ps[i] = e; sm += e; }
    inv_s = 1.f / sm;
  }
  __syncthreads();
  float sc = scale_p[0];
  for (int d = t; d < DMODEL; d += 256) {
    float a2 = 0.f;
    for (int i = 0; i < 32; i++) a2 += ps[i] * fb[(size_t)i * DMODEL + d];
    outp[row * DMODEL + d] = xr[d] + sc * a2 * inv_s;
  }
}

// ---------------- LayerNorm: f32 in -> f16 out ----------------
__global__ __launch_bounds__(256) void k_ln(const float* __restrict__ inp,
                                            const float* __restrict__ g,
                                            const float* __restrict__ bta,
                                            _Float16* __restrict__ outp) {
  __shared__ float r1[256], r2[256];
  int t = threadIdx.x;
  size_t row = blockIdx.x;
  const float* p = inp + row * DMODEL;
  float v0 = p[t], v1 = p[t + 256], v2 = p[t + 512];
  r1[t] = v0 + v1 + v2;
  r2[t] = v0 * v0 + v1 * v1 + v2 * v2;
  __syncthreads();
  for (int s2 = 128; s2 > 0; s2 >>= 1) { if (t < s2) { r1[t] += r1[t + s2]; r2[t] += r2[t + s2]; } __syncthreads(); }
  float mu = r1[0] * (1.f / DMODEL);
  float var = r2[0] * (1.f / DMODEL) - mu * mu;
  float rstd = rsqrtf(var + 1e-6f);
  _Float16* o = outp + row * DMODEL;
  o[t]       = (_Float16)((v0 - mu) * rstd * g[t]       + bta[t]);
  o[t + 256] = (_Float16)((v1 - mu) * rstd * g[t + 256] + bta[t + 256]);
  o[t + 512] = (_Float16)((v2 - mu) * rstd * g[t + 512] + bta[t + 512]);
}

// ---------------- WMMA GEMM: C = A(f16,MxK) @ Bt^T (Bt = f16 [N][K]) + bias ----------------
// Async double-buffered tile pipeline (GLOBAL_LOAD_ASYNC_TO_LDS_B128 + s_wait_asynccnt).
// EPI 0: -> f16 out.  EPI 1: exact GELU -> f16 out.  EPI 2: + res (f32 stream) -> f32 out.
template <int EPI>
__global__ __launch_bounds__(256) void k_gemm(const _Float16* __restrict__ A,
                                              const _Float16* __restrict__ Bt,
                                              const float* __restrict__ bias,
                                              const float* __restrict__ res,
                                              float* __restrict__ outf,
                                              _Float16* __restrict__ outh,
                                              int M, int N, int K) {
  constexpr int AS = 48;                 // half stride -> 96B rows (16B-aligned frags)
  __shared__ alignas(16) _Float16 As[2][128 * AS];
  __shared__ alignas(16) _Float16 Ws[2][64 * AS];
  int tid = threadIdx.x, lane = tid & 31, wv = tid >> 5;
  int gm0 = blockIdx.y * 128, gn0 = blockIdx.x * 64;
  v8f acc[4];
  for (int i = 0; i < 4; i++) for (int e = 0; e < 8; e++) acc[i][e] = 0.f;

  // A: 128 rows x 32 halfs per tile; 2 threads/row, 16 halfs (2x b128) each
  int ar = tid >> 1, apart = (tid & 1) * 16;
  int arow = gm0 + ar; if (arow >= M) arow = M - 1;   // clamp: invalid rows never stored
  const _Float16* aptr = A + (size_t)arow * K + apart;
  unsigned aoff0 = lds_off(&As[0][ar * AS + apart]);
  unsigned aoff1 = lds_off(&As[1][ar * AS + apart]);
  // B: 64 rows x 32 halfs per tile; 4 threads/row, 8 halfs (1x b128) each
  int bn = tid >> 2, bpart = (tid & 3) * 8;
  const _Float16* bptr = Bt + (size_t)(gn0 + bn) * K + bpart;
  unsigned boff0 = lds_off(&Ws[0][bn * AS + bpart]);
  unsigned boff1 = lds_off(&Ws[1][bn * AS + bpart]);

  auto issue = [&](int buf, int kk) {
    unsigned ao = buf ? aoff1 : aoff0;
    unsigned bo = buf ? boff1 : boff0;
    async_b128(ao,      (const void*)(aptr + kk));
    async_b128(ao + 16, (const void*)(aptr + kk + 8));
    async_b128(bo,      (const void*)(bptr + kk));
  };

  int nk = K / 32;
  issue(0, 0);
  for (int i = 0; i < nk; i++) {
    int cur = i & 1;
    wait_async0();          // my tile landed in LDS
    __syncthreads();        // everyone's tile landed; nobody still reads buf cur^1
    if (i + 1 < nk) issue(cur ^ 1, (i + 1) * 32);
    int r = wv * 16 + (lane & 15);
    v16h a = load_a_frag(&As[cur][r * AS], lane);
    for (int nf = 0; nf < 4; nf++) {
      int c = nf * 16 + (lane & 15);
      v16h bf = load_b_frag(&Ws[cur][c * AS], lane);
      acc[nf] = __builtin_amdgcn_wmma_f32_16x16x32_f16(false, a, false, bf, (short)0, acc[nf], false, false);
    }
  }
  int rbase = gm0 + wv * 16 + ((lane < 16) ? 0 : 8);
  for (int nf = 0; nf < 4; nf++) {
    int col = gn0 + nf * 16 + (lane & 15);
    float bv = bias[col];
    for (int v = 0; v < 8; v++) {
      int row = rbase + v;
      if (row >= M) continue;
      float val = acc[nf][v] + bv;
      if (EPI == 1) val = 0.5f * val * (1.f + erff(val * 0.70710678118654752f));
      if (EPI == 2) {
        outf[(size_t)row * N + col] = res[(size_t)row * N + col] + val;
      } else {
        outh[(size_t)row * N + col] = (_Float16)val;
      }
    }
  }
}

// ---------------- WMMA multi-head attention (12 heads, HD=64) ----------------
// qkv: f16 [B*197, 2304], cols: q=h*64, k=768+h*64, v=1536+h*64. out: f16 [B*197, 768]
__global__ __launch_bounds__(64) void k_attn(const _Float16* __restrict__ qkv,
                                             _Float16* __restrict__ outp) {
  constexpr int QS = 72, VS = 40, SW = 224;
  __shared__ alignas(16) _Float16 Qs[32 * QS];
  __shared__ alignas(16) _Float16 Ks[32 * QS];
  __shared__ alignas(16) _Float16 Vt[64 * VS];   // Vt[d][key]
  __shared__ float SS[32 * SW];
  __shared__ float rsum[32];
  int tid = threadIdx.x, lane = tid & 31, wv = tid >> 5;
  int bid = blockIdx.x;
  int mt = bid % 7;
  int h  = (bid / 7) % NHEAD;
  int b  = bid / (7 * NHEAD);
  size_t base = (size_t)b * SEQ * QKVN;

  // async tile stage: 32 rows x 64 halfs = 256 b128 chunks; 4 chunks per thread
  auto issue_tile = [&](const _Float16* gcol, _Float16* lbase, int row0) {
    unsigned lb = lds_off(lbase);
    for (int q = 0; q < 4; q++) {
      int c = tid + 64 * q;
      int rr = c >> 3;
      int off = (c & 7) * 8;
      int n = row0 + rr; if (n >= SEQ) n = SEQ - 1;     // clamp; masked/zero-P later
      async_b128(lb + (unsigned)(rr * QS + off) * 2,
                 (const void*)(gcol + (size_t)n * QKVN + off));
    }
  };

  issue_tile(qkv + base + h * HDIM, Qs, mt * 32);        // Q (unscaled; scale at score write)
  for (int kt = 0; kt < 7; kt++) {                        // S = Q K^T
    int j0 = kt * 32;
    issue_tile(qkv + base + DMODEL + h * HDIM, Ks, j0);
    wait_async0();
    __syncthreads();
    v8f sa[2];
    for (int i = 0; i < 2; i++) for (int e = 0; e < 8; e++) sa[i][e] = 0.f;
    int r = wv * 16 + (lane & 15);
    for (int ks = 0; ks < 2; ks++) {
      v16h a = load_a_frag(Qs + r * QS + ks * 32, lane);
      for (int nf = 0; nf < 2; nf++) {
        int c = nf * 16 + (lane & 15);
        v16h bf = load_b_frag(Ks + c * QS + ks * 32, lane);
        sa[nf] = __builtin_amdgcn_wmma_f32_16x16x32_f16(false, a, false, bf, (short)0, sa[nf], false, false);
      }
    }
    int irow = wv * 16 + ((lane < 16) ? 0 : 8);
    for (int nf = 0; nf < 2; nf++) {
      int jl = nf * 16 + (lane & 15);
      int j = j0 + jl;
      for (int v = 0; v < 8; v++)
        SS[(irow + v) * SW + j0 + jl] = (j < SEQ) ? sa[nf][v] * 0.125f : -1e30f;
    }
    __syncthreads();
  }
  if (tid < 32) {                         // softmax rows (mask >=197; zero pad cols)
    int i = tid;
    float mx = -1e30f;
    for (int j = 0; j < SEQ; j++) mx = fmaxf(mx, SS[i * SW + j]);
    float sm = 0.f;
    for (int j = 0; j < SEQ; j++) { float e = __expf(SS[i * SW + j] - mx); SS[i * SW + j] = e; sm += e; }
    for (int j = SEQ; j < SW; j++) SS[i * SW + j] = 0.f;
    rsum[i] = sm;
  }
  __syncthreads();
  v8f oa[4];
  for (int i = 0; i < 4; i++) for (int e = 0; e < 8; e++) oa[i][e] = 0.f;
  int li = tid >> 1, lpart = (tid & 1) * 32;
  for (int kt = 0; kt < 7; kt++) {        // O = P V
    int j0 = kt * 32;
    {   // V transposed into LDS (manual: element scatter)
      int j = j0 + li;
      if (j < SEQ) {
        const _Float16* sp = qkv + base + (size_t)j * QKVN + 2 * DMODEL + h * HDIM + lpart;
        for (int e = 0; e < 32; e++) Vt[(lpart + e) * VS + li] = sp[e];
      } else for (int e = 0; e < 32; e++) Vt[(lpart + e) * VS + li] = (_Float16)0.f;
    }
    __syncthreads();
    int r = wv * 16 + (lane & 15);
    int kb = (lane < 16) ? 0 : 8;
    const float* sp = SS + r * SW + j0;
    v16h a;
    for (int e = 0; e < 8; e++) { a[e] = (_Float16)sp[kb + e]; a[e + 8] = (_Float16)sp[16 + kb + e]; }
    for (int nf = 0; nf < 4; nf++) {
      int c = nf * 16 + (lane & 15);
      v16h bf = load_b_frag(Vt + c * VS, lane);
      oa[nf] = __builtin_amdgcn_wmma_f32_16x16x32_f16(false, a, false, bf, (short)0, oa[nf], false, false);
    }
    __syncthreads();
  }
  int irow = wv * 16 + ((lane < 16) ? 0 : 8);
  for (int nf = 0; nf < 4; nf++) {
    int dcol = nf * 16 + (lane & 15);
    for (int v = 0; v < 8; v++) {
      int i = irow + v;
      int n = mt * 32 + i;
      if (n < SEQ)
        outp[((size_t)b * SEQ + n) * DMODEL + h * HDIM + dcol] = (_Float16)(oa[nf][v] / rsum[i]);
    }
  }
}

// ---------------- host ----------------
extern "C" void kernel_launch(void* const* d_in, const int* in_sizes, int n_in,
                              void* d_out, int out_size, void* d_ws, size_t ws_size,
                              hipStream_t stream) {
  (void)in_sizes; (void)n_in; (void)out_size; (void)ws_size;
  const float* x       = (const float*)d_in[0];
  const float* y       = (const float*)d_in[1];
  const float* lat     = (const float*)d_in[2];
  const float* scale_a = (const float*)d_in[3];
  const float* scale_v = (const float*)d_in[4];
  const float* const* P  = (const float* const*)(d_in + 5);   // p_* params (12)
  const float* const* Mm = (const float* const*)(d_in + 17);  // m_* params (12)

  char* ws = (char*)d_ws;
  float* xb = (float*)ws;            ws += (size_t)MROWS * DMODEL * 4;
  float* yb = (float*)ws;            ws += (size_t)MROWS * DMODEL * 4;
  float* fused = (float*)ws;         ws += (size_t)BATCH * NLAT * DMODEL * 4;
  _Float16* lnbuf = (_Float16*)ws;   ws += (size_t)MROWS * DMODEL * 2;
  _Float16* attnbuf = (_Float16*)ws; ws += (size_t)MROWS * DMODEL * 2;
  _Float16* bigbuf = (_Float16*)ws;  ws += (size_t)MROWS * FFDIM * 2;   // qkv / mlp-hidden
  _Float16* wtbase = (_Float16*)ws;  // transposed f16 weights: 2 * 7,077,888 halfs

  // per-modality transposed weight slabs
  const size_t SZ_QKV = (size_t)DMODEL * QKVN;   // 1,769,472
  const size_t SZ_PRJ = (size_t)DMODEL * DMODEL; //   589,824
  const size_t SZ_FC  = (size_t)DMODEL * FFDIM;  // 2,359,296
  _Float16* wt[2][4];
  for (int mod = 0; mod < 2; mod++) {
    _Float16* p = wtbase + (size_t)mod * (SZ_QKV + SZ_PRJ + 2 * SZ_FC);
    wt[mod][0] = p;                     // qkv_w^T  [2304][768]
    wt[mod][1] = p + SZ_QKV;            // proj_w^T [768][768]
    wt[mod][2] = p + SZ_QKV + SZ_PRJ;   // fc1_w^T  [3072][768]
    wt[mod][3] = p + SZ_QKV + SZ_PRJ + SZ_FC; // fc2_w^T [768][3072]
  }
  for (int mod = 0; mod < 2; mod++) {
    const float* const* Wp = mod ? Mm : P;
    k_wt<<<dim3(QKVN / 64,  DMODEL / 64), 256, 0, stream>>>(Wp[2],  wt[mod][0], DMODEL, QKVN);
    k_wt<<<dim3(DMODEL / 64, DMODEL / 64), 256, 0, stream>>>(Wp[4],  wt[mod][1], DMODEL, DMODEL);
    k_wt<<<dim3(FFDIM / 64,  DMODEL / 64), 256, 0, stream>>>(Wp[8],  wt[mod][2], DMODEL, FFDIM);
    k_wt<<<dim3(DMODEL / 64, FFDIM / 64),  256, 0, stream>>>(Wp[10], wt[mod][3], FFDIM, DMODEL);
  }

  k_fused<<<BATCH * NLAT, 256, 0, stream>>>(x, y, lat, fused);
  k_cross<<<MROWS, 256, 0, stream>>>(x, fused, scale_a, xb);
  k_cross<<<MROWS, 256, 0, stream>>>(y, fused, scale_v, yb);

  float* outx = (float*)d_out;
  float* outy = outx + (size_t)MROWS * DMODEL;
  dim3 gq(QKVN / 64, (MROWS + 127) / 128);
  dim3 gp(DMODEL / 64, (MROWS + 127) / 128);
  dim3 gf(FFDIM / 64, (MROWS + 127) / 128);

  for (int mod = 0; mod < 2; mod++) {
    const float* const* Wp = mod ? Mm : P;
    float* s = mod ? yb : xb;
    float* op = mod ? outy : outx;
    k_ln<<<MROWS, 256, 0, stream>>>(s, Wp[0], Wp[1], lnbuf);
    k_gemm<0><<<gq, 256, 0, stream>>>(lnbuf, wt[mod][0], Wp[3], nullptr, nullptr, bigbuf, MROWS, QKVN, DMODEL);
    k_attn<<<7 * NHEAD * BATCH, 64, 0, stream>>>(bigbuf, attnbuf);
    k_gemm<2><<<gp, 256, 0, stream>>>(attnbuf, wt[mod][1], Wp[5], s, s, nullptr, MROWS, DMODEL, DMODEL);
    k_ln<<<MROWS, 256, 0, stream>>>(s, Wp[6], Wp[7], lnbuf);
    k_gemm<1><<<gf, 256, 0, stream>>>(lnbuf, wt[mod][2], Wp[9], nullptr, nullptr, bigbuf, MROWS, FFDIM, DMODEL);
    k_gemm<2><<<gp, 256, 0, stream>>>(bigbuf, wt[mod][3], Wp[11], s, op, nullptr, MROWS, DMODEL, FFDIM);
  }
}
